// MambaLayer_11708080849423
// MI455X (gfx1250) — matrix operand order repaired and checked by reading, not compiled, not run on hardware
//
#include <hip/hip_runtime.h>
#include <math.h>

// ---------------------------------------------------------------------------
// Mamba block forward for MI455X (gfx1250, wave32, WMMA + async global->LDS).
// DM=1024, DS=64, DC=4, DI=2048, DR=64, B=2, L=1024.
//
// Pipeline:
//   0) one-shot f32->bf16 packs: x, and weights converted+transposed to [N][K]
//   1) xz   = x  @ W_in        (WMMA bf16, async-LDS staged)
//   2) xi   = silu(causal depthwise conv(xz[:, :DI]))       (f32 + bf16 out)
//   3) proj = xi @ W_x         (WMMA)
//   4) dt   = softplus(proj[:, :DR] @ W_dt + b_dt)          (WMMA + epilogue)
//   5) selective scan + D-skip + silu(z) gate               (VALU, bf16 out)
//   6) out  = y_gated @ W_out  (WMMA)
// ---------------------------------------------------------------------------

#define DM   1024
#define DS   64
#define DC   4
#define DI   2048
#define DR   64
#define BB   2
#define LL   1024
#define NROW (BB * LL)        // 2048
#define XZW  (2 * DI)         // 4096
#define PJW  (DR + 2 * DS)    // 192

typedef __attribute__((ext_vector_type(16))) __bf16 v16bf;
typedef __attribute__((ext_vector_type(8)))  float  v8f;
typedef unsigned short ushort_t;

__device__ __forceinline__ ushort_t f2bf(float f) {
    union { float f; unsigned u; } v; v.f = f;
    unsigned r = v.u + 0x7fffu + ((v.u >> 16) & 1u);   // round-to-nearest-even
    return (ushort_t)(r >> 16);
}

__device__ __forceinline__ float silu_f(float x) {
    return x / (1.0f + __expf(-x));
}

// one async 16-byte global->LDS transfer (GV mode), tracked by ASYNCcnt
__device__ __forceinline__ void async_copy_b128(unsigned lds_off, const void* gptr) {
    asm volatile("global_load_async_to_lds_b128 %0, %1, off"
                 :: "v"(lds_off), "v"((unsigned long long)gptr)
                 : "memory");
}

__device__ __forceinline__ void wait_async_all() {
    asm volatile("s_wait_asynccnt 0x0" ::: "memory");
}

// ---------------------------------------------------------------------------
// One-shot conversion kernels (f32 -> packed bf16).
// ---------------------------------------------------------------------------
__global__ __launch_bounds__(256) void cvt_bf16_kernel(
    const float* __restrict__ in, ushort_t* __restrict__ out)
{
    size_t i = (size_t)blockIdx.x * 256 + threadIdx.x;
    out[i] = f2bf(in[i]);
}

// W[K][N] f32  ->  Wt[N][K] bf16   (weights: convert + transpose once)
__global__ __launch_bounds__(256) void cvtT_bf16_kernel(
    const float* __restrict__ W, ushort_t* __restrict__ Wt, int K, int N)
{
    size_t i = (size_t)blockIdx.x * 256 + threadIdx.x;   // over K*N
    int n = (int)(i % N);
    int k = (int)(i / N);
    Wt[(size_t)n * K + k] = f2bf(W[i]);                  // coalesced read
}

// strided slice convert: out[r][c] = bf16(in[r*ld + c]), c < cols
__global__ __launch_bounds__(256) void cvt_slice_kernel(
    const float* __restrict__ in, ushort_t* __restrict__ out, int cols, int ld)
{
    size_t i = (size_t)blockIdx.x * 256 + threadIdx.x;
    int c = (int)(i % cols);
    int r = (int)(i / cols);
    out[i] = f2bf(in[(size_t)r * ld + c]);
}

// ---------------------------------------------------------------------------
// Tiled bf16 WMMA GEMM:  C[M x N] = epilogue(A[M x K] @ B[K x N])
// A: bf16 [M][lda] row-major.  Bt: bf16 [N][ldb] (B pre-transposed, K-major).
// Tiles staged into LDS with GLOBAL_LOAD_ASYNC_TO_LDS_B128 (no VGPR transit).
// Block tile BM x BN, BK=32; one wave per 16-wide N slab, BM/16 WMMAs/chunk.
// Fragment packing per ISA 7.12.2 wave32 layouts (aligned b32 LDS reads).
// EPI: 0 = none, 1 = softplus(acc + bias[n])
// ---------------------------------------------------------------------------
template <int BM, int BN, int EPI>
__global__ __launch_bounds__(BN * 2) void gemm_bf16_wmma(
    const ushort_t* __restrict__ A, const ushort_t* __restrict__ Bt,
    float* __restrict__ C, const float* __restrict__ bias,
    int M, int N, int K, int lda, int ldb, int ldc)
{
    constexpr int NT = BN * 2;                       // (BN/16) waves * 32
    __shared__ __align__(16) ushort_t sA[BM * 32];   // [BM][32] row-major
    __shared__ __align__(16) ushort_t sBt[BN * 32];  // [BN][32] K-major

    const int tid  = threadIdx.x;
    const int wave = tid >> 5;
    const int lane = tid & 31;
    const int half = lane >> 4;
    const int lr   = lane & 15;
    const int gm0  = blockIdx.y * BM;
    const int gn0  = blockIdx.x * BN;

    const unsigned sA_base  = (unsigned)(size_t)(const void*)sA;   // LDS offset
    const unsigned sBt_base = (unsigned)(size_t)(const void*)sBt;

    v8f acc[BM / 16] = {};

    for (int kb = 0; kb < K; kb += 32) {
        // async-stage A tile: BM rows x 32 bf16 = BM*4 x 16B chunks
        for (int c = tid; c < BM * 4; c += NT) {
            int row = c >> 2, part = c & 3;
            async_copy_b128(sA_base + c * 16,
                            A + (size_t)(gm0 + row) * lda + kb + part * 8);
        }
        // async-stage Bt tile: BN rows x 32 bf16 = BN*4 x 16B chunks
        for (int c = tid; c < BN * 4; c += NT) {
            int n = c >> 2, part = c & 3;
            async_copy_b128(sBt_base + c * 16,
                            Bt + (size_t)(gn0 + n) * ldb + kb + part * 8);
        }
        wait_async_all();
        __syncthreads();

        // B fragment: lane holds col N = wave*16+lr; dword j = K pair {((j<4)?0:16) + 8*half + 2*(j&3), +1}
        union { v16bf v; unsigned u[8]; } bf;
        #pragma unroll
        for (int j = 0; j < 8; ++j)
            bf.u[j] = *(const unsigned*)&sBt[(wave * 16 + lr) * 32 + ((j & 4) ? 16 : 0) + 8 * half + 2 * (j & 3)];

        #pragma unroll
        for (int mi = 0; mi < BM / 16; ++mi) {
            // A fragment: lane holds row M = mi*16+lr;
            // dword j = K pair {((j<4)?0:16) + 8*half + 2*(j&3), +1}
            union { v16bf v; unsigned u[8]; } af;
            #pragma unroll
            for (int j = 0; j < 8; ++j) {
                int kk = ((j & 4) ? 16 : 0) + 8 * half + 2 * (j & 3);
                af.u[j] = *(const unsigned*)&sA[(mi * 16 + lr) * 32 + kk];
            }
            acc[mi] = __builtin_amdgcn_wmma_f32_16x16x32_bf16(
                false, af.v, false, bf.v, (short)0, acc[mi], false, false);
        }
        __syncthreads();
    }

    // store: VGPR j holds M = j + 8*half, N = lane%16 (C/D layout, ISA 7.12.2)
    const int col = gn0 + wave * 16 + lr;
    #pragma unroll
    for (int mi = 0; mi < BM / 16; ++mi) {
        #pragma unroll
        for (int j = 0; j < 8; ++j) {
            int row = gm0 + mi * 16 + j + 8 * half;
            float v = acc[mi][j];
            if (EPI == 1) {
                v += bias[col];
                v = (v > 20.0f) ? v : log1pf(__expf(v));   // softplus
            }
            C[(size_t)row * ldc + col] = v;
        }
    }
}

// ---------------------------------------------------------------------------
// Depthwise causal conv1d (DC=4 taps, left pad 3) + bias + SiLU.
// Emits f32 (for the scan) and bf16 (A-operand of the x_proj GEMM).
// ---------------------------------------------------------------------------
__global__ __launch_bounds__(256) void conv_silu_kernel(
    const float* __restrict__ xz, const float* __restrict__ cw,
    const float* __restrict__ cb, float* __restrict__ xi_f,
    ushort_t* __restrict__ xi_bf)
{
    size_t i = (size_t)blockIdx.x * 256 + threadIdx.x;   // over B*L*DI
    int d  = (int)(i % DI);
    int bl = (int)(i / DI);
    int l  = bl % LL;
    float s = cb[d];
    #pragma unroll
    for (int k = 0; k < DC; ++k) {
        int ls = l - (DC - 1) + k;
        if (ls >= 0)
            s += xz[((size_t)(bl - l + ls)) * XZW + d] * cw[d * DC + k];
    }
    s = silu_f(s);
    xi_f[i]  = s;
    xi_bf[i] = f2bf(s);
}

// ---------------------------------------------------------------------------
// Selective scan. One lane owns one (b,d) channel: h[64], A[64] in VGPRs.
// B_t/C_t (64 f32 each) staged once per block per timestep through LDS.
// Fuses the D-skip and the silu(z) gate; emits bf16 for the output GEMM.
// ---------------------------------------------------------------------------
__global__ __launch_bounds__(256) void scan_kernel(
    const float* __restrict__ proj,   // (NROW, 192): [dt | B | C]
    const float* __restrict__ dtb,    // (NROW, DI) softplus'd dt
    const float* __restrict__ xi,     // (NROW, DI) conv+silu activations
    const float* __restrict__ xz,     // (NROW, 4096): z = cols [DI, 2*DI)
    const float* __restrict__ A_log,  // (DI, DS)
    const float* __restrict__ Dp,     // (DI)
    ushort_t* __restrict__ yg)        // (NROW, DI) gated output, bf16
{
    const int d = blockIdx.x * 256 + threadIdx.x;   // channel
    const int b = blockIdx.y;                       // batch

    float Av[DS], h[DS];
    #pragma unroll
    for (int n = 0; n < DS; ++n) {
        Av[n] = -__expf(A_log[(size_t)d * DS + n]);
        h[n]  = 0.0f;
    }
    const float Dd = Dp[d];

    __shared__ float sB[DS], sC[DS];

    for (int t = 0; t < LL; ++t) {
        size_t row = (size_t)(b * LL + t);
        if (threadIdx.x < DS)
            sB[threadIdx.x] = proj[row * PJW + DR + threadIdx.x];
        else if (threadIdx.x < 2 * DS)
            sC[threadIdx.x - DS] = proj[row * PJW + DR + DS + (threadIdx.x - DS)];
        __syncthreads();

        const float dtv = dtb[row * DI + d];
        const float xv  = xi[row * DI + d];
        const float dtx = dtv * xv;
        float y = 0.0f;
        #pragma unroll
        for (int n = 0; n < DS; ++n) {
            h[n] = h[n] * __expf(dtv * Av[n]) + dtx * sB[n];
            y += h[n] * sC[n];
        }
        const float zz = xz[row * XZW + DI + d];
        yg[row * DI + d] = f2bf((y + xv * Dd) * silu_f(zz));
        __syncthreads();
    }
}

// ---------------------------------------------------------------------------
// Host-side orchestration (graph-capture safe: launches only).
// ---------------------------------------------------------------------------
extern "C" void kernel_launch(void* const* d_in, const int* in_sizes, int n_in,
                              void* d_out, int out_size, void* d_ws, size_t ws_size,
                              hipStream_t stream)
{
    (void)in_sizes; (void)n_in; (void)out_size; (void)ws_size;

    const float* x      = (const float*)d_in[0];   // (B,L,DM)
    const float* W_in   = (const float*)d_in[1];   // (DM, 2*DI)
    const float* conv_w = (const float*)d_in[2];   // (DI,1,DC)
    const float* conv_b = (const float*)d_in[3];   // (DI)
    const float* W_x    = (const float*)d_in[4];   // (DI, DR+2*DS)
    const float* W_dt   = (const float*)d_in[5];   // (DR, DI)
    const float* b_dt   = (const float*)d_in[6];   // (DI)
    const float* A_log  = (const float*)d_in[7];   // (DI,DS)
    const float* Dp     = (const float*)d_in[8];   // (DI)
    const float* W_out  = (const float*)d_in[9];   // (DI, DM)
    float* out = (float*)d_out;                    // (B,L,DM)

    // ---- workspace partition -------------------------------------------
    float* fws  = (float*)d_ws;
    float* xz   = fws;                              // NROW*4096  f32
    float* xi   = xz + (size_t)NROW * XZW;          // NROW*2048  f32
    float* proj = xi + (size_t)NROW * DI;           // NROW*192   f32
    float* dtb  = proj + (size_t)NROW * PJW;        // NROW*2048  f32

    ushort_t* hws   = (ushort_t*)(dtb + (size_t)NROW * DI);
    ushort_t* xbf   = hws;                          // NROW*1024
    ushort_t* xibf  = xbf  + (size_t)NROW * DM;     // NROW*2048
    ushort_t* dtraw = xibf + (size_t)NROW * DI;     // NROW*64
    ushort_t* ygbf  = dtraw + (size_t)NROW * DR;    // NROW*2048
    ushort_t* WtIn  = ygbf  + (size_t)NROW * DI;    // [4096][1024]
    ushort_t* WtX   = WtIn  + (size_t)XZW * DM;     // [192][2048]
    ushort_t* WtDt  = WtX   + (size_t)PJW * DI;     // [2048][64]
    ushort_t* WtOut = WtDt  + (size_t)DI * DR;      // [1024][2048]

    // ---- 0) one-shot bf16 packs ----------------------------------------
    cvt_bf16_kernel<<<(size_t)NROW * DM / 256, 256, 0, stream>>>(x, xbf);
    cvtT_bf16_kernel<<<(size_t)DM * XZW / 256, 256, 0, stream>>>(W_in, WtIn, DM, XZW);
    cvtT_bf16_kernel<<<(size_t)DI * PJW / 256, 256, 0, stream>>>(W_x, WtX, DI, PJW);
    cvtT_bf16_kernel<<<(size_t)DR * DI  / 256, 256, 0, stream>>>(W_dt, WtDt, DR, DI);
    cvtT_bf16_kernel<<<(size_t)DI * DM  / 256, 256, 0, stream>>>(W_out, WtOut, DI, DM);

    // ---- 1) xz = x @ W_in            (2048 x 1024) @ (1024 x 4096) -----
    gemm_bf16_wmma<64, 128, 0><<<dim3(XZW / 128, NROW / 64), 256, 0, stream>>>(
        xbf, WtIn, xz, nullptr, NROW, XZW, DM, DM, DM, XZW);

    // ---- 2) depthwise causal conv + SiLU -------------------------------
    conv_silu_kernel<<<(size_t)NROW * DI / 256, 256, 0, stream>>>(
        xz, conv_w, conv_b, xi, xibf);

    // ---- 3) proj = xi @ W_x          (2048 x 2048) @ (2048 x 192) ------
    gemm_bf16_wmma<64, 64, 0><<<dim3(PJW / 64, NROW / 64), 128, 0, stream>>>(
        xibf, WtX, proj, nullptr, NROW, PJW, DI, DI, DI, PJW);

    // ---- 4) dt = softplus(proj[:, :64] @ W_dt + b_dt) ------------------
    cvt_slice_kernel<<<(size_t)NROW * DR / 256, 256, 0, stream>>>(
        proj, dtraw, DR, PJW);
    gemm_bf16_wmma<64, 128, 1><<<dim3(DI / 128, NROW / 64), 256, 0, stream>>>(
        dtraw, WtDt, dtb, b_dt, NROW, DI, DR, DR, DR, DI);

    // ---- 5) selective scan + D skip + silu(z) gate ---------------------
    scan_kernel<<<dim3(DI / 256, BB), 256, 0, stream>>>(
        proj, dtb, xi, xz, A_log, Dp, ygbf);

    // ---- 6) out = y_gated @ W_out    (2048 x 2048) @ (2048 x 1024) -----
    gemm_bf16_wmma<64, 128, 0><<<dim3(DM / 128, NROW / 64), 256, 0, stream>>>(
        ygbf, WtOut, out, nullptr, NROW, DM, DI, DI, DI, DM);
}
